// RotatE_20873541059112
// MI455X (gfx1250) — matrix operand order repaired and checked by reading, not compiled, not run on hardware
//
#include <hip/hip_runtime.h>

#define N_ENT   100000
#define N_REL   500
#define DIM     1024
#define BATCH   1024
#define KDIM    (2 * DIM)          // 2048: [re | im] along K

typedef __attribute__((ext_vector_type(16))) __bf16 v16bf;
typedef __attribute__((ext_vector_type(8)))  float  v8f;

union FragU { v16bf v; uint4 q[2]; };          // 32 B bf16 fragment
union Pk4  { __bf16 h[4]; uint2 u; };          // 4 packed bf16

#if defined(__HIP_DEVICE_COMPILE__) && \
    __has_builtin(__builtin_amdgcn_global_load_async_to_lds_b128)
#define USE_ASYNC_LDS 1
typedef int v4i __attribute__((vector_size(4 * sizeof(int))));
typedef __attribute__((address_space(1))) v4i* g_v4i;   // global (AS1)
typedef __attribute__((address_space(3))) v4i* l_v4i;   // LDS (AS3)
#else
#define USE_ASYNC_LDS 0
#endif

// ---------------------------------------------------------------------------
// Kernel 1: gather + complex rotation, write Q^T in bf16 to workspace.
//   qt[k][b], k in [0,2048): k<1024 -> q_re(d=k), k>=1024 -> q_im(d=k-1024)
// Threads: 1M total; within a block d is fixed, b consecutive -> coalesced.
// ---------------------------------------------------------------------------
__global__ void rotate_qt_kernel(const int* __restrict__ e1,
                                 const int* __restrict__ r,
                                 const float* __restrict__ ent_re,
                                 const float* __restrict__ ent_im,
                                 const float* __restrict__ rel_phase,
                                 __bf16* __restrict__ qt) {
    int idx = blockIdx.x * blockDim.x + threadIdx.x;   // 0 .. 1M-1
    int d = idx >> 10;
    int b = idx & (BATCH - 1);
    int e  = e1[b];
    int rl = r[b];
    float h_re = ent_re[(size_t)e * DIM + d];
    float h_im = ent_im[(size_t)e * DIM + d];
    float ph   = rel_phase[(size_t)rl * DIM + d];
    float s, c;
    __sincosf(ph, &s, &c);
    float q_re = h_re * c - h_im * s;
    float q_im = h_re * s + h_im * c;
    qt[(size_t)d * BATCH + b]         = (__bf16)q_re;
    qt[(size_t)(DIM + d) * BATCH + b] = (__bf16)q_im;
}

// ---------------------------------------------------------------------------
// Kernel 2: score[b, n] = sum_k Qt[k][b] * E[n][k]   (E = [ent_re|ent_im])
// A = E tile (M = n), B = Qt tile (N = b). WMMA f32_16x16x32_bf16.
// Workgroup: 256 thr = 8 waves -> 128(n) x 128(b) tile; wave = 4(n) x 2(b)
// sub-tiles of 16x16. K loop step 32, fp32->bf16 convert on E stage-in.
// Qt tile staged with async global->LDS (CDNA5 ASYNCcnt path).
// ---------------------------------------------------------------------------
#define EA_PITCH 40    // bf16 elems per row (32 + pad), 80 B: 16B-aligned chunks
#define QB_PITCH 136   // bf16 elems per row (128 + pad), 272 B

__global__ __launch_bounds__(256)
void rotate_score_gemm(const __bf16* __restrict__ qt,
                       const float* __restrict__ ent_re,
                       const float* __restrict__ ent_im,
                       float* __restrict__ out) {
    __shared__ __align__(16) __bf16 Ea[128 * EA_PITCH];  // [n][k] bf16
    __shared__ __align__(16) __bf16 Qb[32 * QB_PITCH];   // [k][b] bf16

    const int tid  = threadIdx.x;
    const int lane = tid & 31;
    const int wid  = tid >> 5;
    const int wn   = wid & 1;          // 0..1 over n
    const int wb   = wid >> 1;         // 0..3 over b

    const int b_base = blockIdx.x * 128;   // 8 tiles cover BATCH
    const int n_base = blockIdx.y * 128;   // 782 tiles cover N_ENT

    v8f acc[4][2];
#pragma unroll
    for (int i = 0; i < 4; ++i)
#pragma unroll
        for (int j = 0; j < 2; ++j)
            acc[i][j] = (v8f){0.f, 0.f, 0.f, 0.f, 0.f, 0.f, 0.f, 0.f};

    // Per-lane A-fragment addressing (16-bit A 16x32 layout):
    //   lanes 0-15: M=lane, K chunks [0,8) and [16,24)
    //   lanes 16-31: M=lane-16, K chunks [8,16) and [24,32)
    const int a_m  = lane & 15;
    const int a_k0 = (lane >> 4) * 8;

    for (int kb = 0; kb < KDIM; kb += 32) {
        const float* eptr = (kb < DIM) ? ent_re : ent_im;
        const int koff = kb & (DIM - 1);

        __syncthreads();   // all waves done reading previous LDS tiles

        // ---- stage Qt tile: 32 k x 128 b, already bf16: async copy to LDS ----
        {
            int kk  = tid >> 3;            // 256 chunks of 16 bf16
            int seg = tid & 7;
            const __bf16* gsrc = qt + (size_t)(kb + kk) * BATCH + b_base + seg * 16;
            __bf16*       ldst = &Qb[kk * QB_PITCH + seg * 16];
#if USE_ASYNC_LDS
            __builtin_amdgcn_global_load_async_to_lds_b128(
                (g_v4i)gsrc, (l_v4i)ldst, 0, 0);
            __builtin_amdgcn_global_load_async_to_lds_b128(
                (g_v4i)gsrc, (l_v4i)ldst, 16, 0);   // inst_offset hits both addrs
#else
            const uint4* g = (const uint4*)gsrc;
            uint4 q0 = g[0];
            uint4 q1 = g[1];
            uint4* dst = (uint4*)ldst;
            dst[0] = q0;
            dst[1] = q1;
#endif
        }

        // ---- stage E tile: 128 rows x 32 k, fp32 -> bf16 via VGPRs ----
        const int kb2 = kb + 32;
        const float* eptr2 = (kb2 < DIM) ? ent_re : ent_im;
        const int koff2 = kb2 & (DIM - 1);
#pragma unroll
        for (int i = 0; i < 4; ++i) {
            int c   = tid + 256 * i;       // 1024 chunks of 4 floats
            int row = c >> 3;
            int seg = c & 7;
            int rg  = n_base + row;
            if (rg >= N_ENT) rg = N_ENT - 1;
            float4 f = *(const float4*)(eptr + (size_t)rg * DIM + koff + seg * 4);
            if (kb2 < KDIM)   // prefetch next K-slab (global_prefetch_b8)
                __builtin_prefetch(eptr2 + (size_t)rg * DIM + koff2 + seg * 4, 0, 1);
            Pk4 p;
            p.h[0] = (__bf16)f.x; p.h[1] = (__bf16)f.y;
            p.h[2] = (__bf16)f.z; p.h[3] = (__bf16)f.w;
            *(uint2*)&Ea[row * EA_PITCH + seg * 4] = p.u;
        }

#if USE_ASYNC_LDS
#if __has_builtin(__builtin_amdgcn_s_wait_asynccnt)
        __builtin_amdgcn_s_wait_asynccnt(0);   // async LDS writes landed
#else
        asm volatile("s_wait_asynccnt 0x0" ::: "memory");
#endif
#endif
        __syncthreads();

        // ---- fragments + WMMA ----
        FragU fa[4];
#pragma unroll
        for (int i = 0; i < 4; ++i) {
            int n_row = wn * 64 + i * 16 + a_m;
            const __bf16* base = &Ea[n_row * EA_PITCH + a_k0];
            fa[i].q[0] = *(const uint4*)(base);
            fa[i].q[1] = *(const uint4*)(base + 16);
        }
        FragU fb[2];
#pragma unroll
        for (int j = 0; j < 2; ++j) {
            int tb = wb * 32 + j * 16;      // lane%16 = b column, elements = K (table order)
            const __bf16* pb = &Qb[a_k0 * QB_PITCH + tb + a_m];
#pragma unroll
            for (int s = 0; s < 8; ++s) {
                fb[j].v[s]     = pb[s * QB_PITCH];
                fb[j].v[8 + s] = pb[(16 + s) * QB_PITCH];
            }
        }
#pragma unroll
        for (int i = 0; i < 4; ++i)
#pragma unroll
            for (int j = 0; j < 2; ++j)
                acc[i][j] = __builtin_amdgcn_wmma_f32_16x16x32_bf16(
                    false, fa[i].v, false, fb[j].v,
                    (short)0, acc[i][j], false, false);
    }

    // ---- store: C lanes = b (N dim), VGPRs = n (M dim) ----
    const int c_b   = lane & 15;
    const int c_row = (lane >> 4) * 8;     // rows n0+[0..7] or n0+[8..15]
#pragma unroll
    for (int i = 0; i < 4; ++i) {
        int n_t = n_base + wn * 64 + i * 16;
        if (n_t >= N_ENT) continue;        // N_ENT % 16 == 0: whole sub-tile OOB
#pragma unroll
        for (int j = 0; j < 2; ++j) {
            int b_t = b_base + wb * 32 + j * 16;
            float* po = out + (size_t)(b_t + c_b) * N_ENT + n_t + c_row;
            float4 lo = {acc[i][j][0], acc[i][j][1], acc[i][j][2], acc[i][j][3]};
            float4 hi = {acc[i][j][4], acc[i][j][5], acc[i][j][6], acc[i][j][7]};
            *(float4*)(po)     = lo;
            *(float4*)(po + 4) = hi;
        }
    }
}

// ---------------------------------------------------------------------------
extern "C" void kernel_launch(void* const* d_in, const int* in_sizes, int n_in,
                              void* d_out, int out_size, void* d_ws, size_t ws_size,
                              hipStream_t stream) {
    const int*   e1        = (const int*)d_in[0];
    const int*   r         = (const int*)d_in[1];
    const float* ent_re    = (const float*)d_in[2];
    const float* ent_im    = (const float*)d_in[3];
    const float* rel_phase = (const float*)d_in[4];
    float*  out = (float*)d_out;
    __bf16* qt  = (__bf16*)d_ws;           // 2048 x 1024 bf16 = 4 MB

    rotate_qt_kernel<<<(BATCH * DIM) / 256, 256, 0, stream>>>(
        e1, r, ent_re, ent_im, rel_phase, qt);

    dim3 grid(BATCH / 128, (N_ENT + 127) / 128);   // (8, 782)
    rotate_score_gemm<<<grid, 256, 0, stream>>>(qt, ent_re, ent_im, out);
}
